// VQEmbeddingEMA_58428735094911
// MI455X (gfx1250) — matrix operand order repaired and checked by reading, not compiled, not run on hardware
//
#include <hip/hip_runtime.h>

typedef __attribute__((ext_vector_type(16))) _Float16 v16h;
typedef __attribute__((ext_vector_type(8)))  _Float16 v8h;
typedef __attribute__((ext_vector_type(8)))  float    v8f;

#define N_CB        4
#define M_CODES     1024
#define D_DIM       64
#define L_LAT       16
#define CHUNK_M     256
#define ES_STRIDE   72      // halves per code row in LDS (padded vs 64)
#define XS_STRIDE   65      // floats per x row in LDS (padded vs 64)
#define DECAY       0.99f
#define ONE_M_DECAY 0.01f
#define EPS_F       1e-5f
#define BETA_F      0.25f

// ---------------------------------------------------------------------------
// K0: init accumulators. new_ema_weight region starts as 0.99*ema_weight and
// receives 0.01*x scatter-adds from the main kernel (== 0.99*w + 0.01*dw).
// ---------------------------------------------------------------------------
__global__ void vq_init_kernel(const float* __restrict__ ema_weight,
                               float* __restrict__ ew_out,
                               float* __restrict__ cnt,
                               float* __restrict__ sse,
                               float* __restrict__ perp) {
  int i = blockIdx.x * blockDim.x + threadIdx.x;
  if (i < N_CB * M_CODES * D_DIM) ew_out[i] = DECAY * ema_weight[i];
  if (i < N_CB * M_CODES)         cnt[i]    = 0.0f;
  if (i == 0) { sse[0] = 0.0f; perp[0] = 0.0f; }
}

// ---------------------------------------------------------------------------
// K1: main WMMA kernel. 256 threads = 8 wave32 waves. Block owns codebook n
// and 128 token rows (16 per wave). Argmin metric is ||e||^2 - 2 x.e
// (||x||^2 is constant per row and cannot change the argmin).
// ---------------------------------------------------------------------------
__global__ __launch_bounds__(256) void vq_main_kernel(
    const float* __restrict__ x,        // [B, N*D*L]
    const float* __restrict__ emb,      // [N, M, D]
    float* __restrict__ zq,             // [B, N*D*L]
    float* __restrict__ enczq,          // same layout as zq
    float* __restrict__ idx_out,        // [B, N, L] as float
    float* __restrict__ cnt_accum,      // [N, M]   (ws)
    float* __restrict__ ew_accum,       // [N, M, D] (d_out new_ema_weight)
    float* __restrict__ sse_accum,      // [1]       (ws)
    int B) {
  extern __shared__ char smem[];
  _Float16* Es = (_Float16*)smem;                                   // 256*72 halves
  float*    esq = (float*)(smem + CHUNK_M * ES_STRIDE * 2);         // 256 f32
  float*    Xs  = (float*)(smem + CHUNK_M * ES_STRIDE * 2 + CHUNK_M * 4); // 8*16*65 f32

  const int T       = B * L_LAT;
  const int tgroups = T / 128;                 // 8 waves * 16 rows per block
  const int n    = blockIdx.x / tgroups;
  const int g    = blockIdx.x % tgroups;
  const int tid  = threadIdx.x;
  const int wave = tid >> 5;
  const int lane = tid & 31;
  const int t0   = (g * 8 + wave) * 16;
  const int NDL  = N_CB * D_DIM * L_LAT;       // 4096

  const float* embN = emb + (size_t)n * M_CODES * D_DIM;
  float* Xw = Xs + wave * 16 * XS_STRIDE;

  // ---- stage X tile (16 rows x 64) into LDS (x_flat[n][t][d], d-stride L) ----
#pragma unroll
  for (int i = 0; i < 32; ++i) {
    int e = lane + i * 32;            // 0..1023
    int row = e >> 6, d = e & 63;
    int t = t0 + row, b = t >> 4, l = t & 15;
    Xw[row * XS_STRIDE + d] = x[(size_t)b * NDL + n * (D_DIM * L_LAT) + d * L_LAT + l];
  }
  __syncthreads();

  // ---- build A fragments (16x32 f16, K-bases 0 and 32) ----
  const int mrow  = lane & 15;        // A-matrix row handled by this lane
  const int half8 = (lane < 16) ? 0 : 8;
  const float* xr = Xw + mrow * XS_STRIDE;
  v16h a0, a1;
#pragma unroll
  for (int j = 0; j < 8; ++j) {
    a0[j]     = (_Float16)xr[half8 + j];
    a0[j + 8] = (_Float16)xr[16 + half8 + j];
    a1[j]     = (_Float16)xr[32 + half8 + j];
    a1[j + 8] = (_Float16)xr[48 + half8 + j];
  }

  float best[8];
  int   bidx[8];
#pragma unroll
  for (int r = 0; r < 8; ++r) { best[r] = 3.4e38f; bidx[r] = 0; }

  const int koffB = (lane < 16) ? 0 : 8;

  // ---- loop over 4 chunks of 256 codes ----
  for (int ci = 0; ci < 4; ++ci) {
    const int c0chunk = ci * CHUNK_M;
    __syncthreads();   // previous chunk fully consumed before overwrite
    // stage embedding chunk as f16 + fp32 norms
    for (int i = 0; i < 64; ++i) {
      int e = tid + i * 256;
      int m = e >> 6, d = e & 63;
      Es[m * ES_STRIDE + d] = (_Float16)embN[(size_t)(c0chunk + m) * D_DIM + d];
    }
    {
      const float* er = embN + (size_t)(c0chunk + tid) * D_DIM;
      float s = 0.0f;
      for (int d = 0; d < D_DIM; ++d) { float v = er[d]; s += v * v; }
      esq[tid] = s;
    }
    if (ci < 3)  // pull next chunk toward the caches
      __builtin_prefetch(embN + (size_t)(c0chunk + CHUNK_M) * D_DIM + tid, 0, 1);
    __syncthreads();

    // 16 code tiles of 16 codes; 2 WMMAs each (K=64); 2-way unroll for two
    // independent accumulation chains (hides WMMA RAW + ds_load latency).
#pragma unroll 2
    for (int tile = 0; tile < 16; ++tile) {
      const int lc = tile * 16 + (lane & 15);
      const _Float16* ep = Es + lc * ES_STRIDE + koffB;
      v8h blo0 = *(const v8h*)(ep);
      v8h bhi0 = *(const v8h*)(ep + 16);
      v8h blo1 = *(const v8h*)(ep + 32);
      v8h bhi1 = *(const v8h*)(ep + 48);
      v16h b0 = __builtin_shufflevector(blo0, bhi0, 0,1,2,3,4,5,6,7,8,9,10,11,12,13,14,15);
      v16h b1 = __builtin_shufflevector(blo1, bhi1, 0,1,2,3,4,5,6,7,8,9,10,11,12,13,14,15);
      v8f c = {};
      c = __builtin_amdgcn_wmma_f32_16x16x32_f16(false, a0, false, b0, (short)0, c, false, false);
      c = __builtin_amdgcn_wmma_f32_16x16x32_f16(false, a1, false, b1, (short)0, c, false, false);
      const float es  = esq[lc];
      const int gidx  = c0chunk + lc;
#pragma unroll
      for (int r = 0; r < 8; ++r) {
        float dd = es - 2.0f * c[r];      // ||e||^2 - 2 x.e  (x_sq row-constant)
        if (dd < best[r]) { best[r] = dd; bidx[r] = gidx; }
      }
    }
  }

  // ---- argmin across the 16 lanes holding the 16 columns (ties -> low idx) ----
#pragma unroll
  for (int r = 0; r < 8; ++r) {
    float d = best[r]; int ix = bidx[r];
#pragma unroll
    for (int mset = 8; mset >= 1; mset >>= 1) {
      float od = __shfl_xor(d, mset, 32);
      int   oi = __shfl_xor(ix, mset, 32);
      if (od < d || (od == d && oi < ix)) { d = od; ix = oi; }
    }
    best[r] = d; bidx[r] = ix;
  }

  // ---- outputs: gather codebook rows, scatter EMA updates, SSE ----
  float sse = 0.0f;
  const int d0 = lane * 2;      // each lane handles 2 of the 64 dims
#pragma unroll
  for (int row = 0; row < 16; ++row) {
    const int ix = __shfl(bidx[row & 7], (row < 8) ? 0 : 16, 32);
    const int t = t0 + row, b = t >> 4, l = t & 15;
    const float* er = embN + (size_t)ix * D_DIM;
    const float q0 = er[d0], q1 = er[d0 + 1];
    const float x0 = Xw[row * XS_STRIDE + d0], x1 = Xw[row * XS_STRIDE + d0 + 1];
    const size_t ob = (size_t)b * NDL + n * (D_DIM * L_LAT) + l;
    zq[ob + d0 * L_LAT]       = q0;  zq[ob + (d0 + 1) * L_LAT]    = q1;
    enczq[ob + d0 * L_LAT]    = q0;  enczq[ob + (d0 + 1) * L_LAT] = q1;
    const float e0 = x0 - q0, e1 = x1 - q1;
    sse += e0 * e0 + e1 * e1;
    float* ewr = ew_accum + ((size_t)n * M_CODES + ix) * D_DIM;
    atomicAdd(&ewr[d0],     ONE_M_DECAY * x0);
    atomicAdd(&ewr[d0 + 1], ONE_M_DECAY * x1);
    if (lane == 0) {
      idx_out[(size_t)b * (N_CB * L_LAT) + n * L_LAT + l] = (float)ix;
      atomicAdd(&cnt_accum[n * M_CODES + ix], 1.0f);
    }
  }
#pragma unroll
  for (int mset = 16; mset >= 1; mset >>= 1) sse += __shfl_xor(sse, mset, 32);
  if (lane == 0) atomicAdd(sse_accum, sse);
}

// ---------------------------------------------------------------------------
// K2: per-n finalize: Laplace smoothing, perplexity, embedding = w / count,
// loss scalar.
// ---------------------------------------------------------------------------
__global__ __launch_bounds__(256) void vq_finalize_kernel(
    const float* __restrict__ ema_count_in,   // [N, M]
    const float* __restrict__ cnt,            // raw counts (ws)
    const float* __restrict__ ew,             // final new_ema_weight (d_out)
    float* __restrict__ count_out,            // new_ema_count (d_out)
    float* __restrict__ emb_out,              // new_embedding (d_out)
    float* __restrict__ perp_out,
    float* __restrict__ loss_out,
    const float* __restrict__ sse,
    int T) {
  __shared__ float scnt[M_CODES];
  __shared__ float red[256];
  const int n = blockIdx.x, tid = threadIdx.x;

  float csum = 0.0f, ent = 0.0f;
  for (int m = tid; m < M_CODES; m += 256) {
    float raw = cnt[n * M_CODES + m];
    float nc  = DECAY * ema_count_in[n * M_CODES + m] + ONE_M_DECAY * raw;
    scnt[m] = nc;
    csum += nc;
    float p = raw / (float)T;
    ent += p * logf(p + 1e-10f);
  }
  red[tid] = csum; __syncthreads();
  for (int s = 128; s > 0; s >>= 1) { if (tid < s) red[tid] += red[tid + s]; __syncthreads(); }
  const float ntot = red[0];
  __syncthreads();
  red[tid] = ent; __syncthreads();
  for (int s = 128; s > 0; s >>= 1) { if (tid < s) red[tid] += red[tid + s]; __syncthreads(); }
  if (tid == 0) atomicAdd(perp_out, expf(-red[0]));
  __syncthreads();

  const float scale = ntot / (ntot + (float)M_CODES * EPS_F);
  for (int m = tid; m < M_CODES; m += 256) {
    float sm = (scnt[m] + EPS_F) * scale;
    count_out[n * M_CODES + m] = sm;
    scnt[m] = sm;
  }
  __syncthreads();
  for (int e = tid; e < M_CODES * D_DIM; e += 256) {
    int m = e >> 6;
    emb_out[(size_t)n * M_CODES * D_DIM + e] =
        ew[(size_t)n * M_CODES * D_DIM + e] / scnt[m];
  }
  if (n == 0 && tid == 0)
    loss_out[0] = BETA_F * sse[0] / (float)((size_t)N_CB * T * D_DIM);
}

// ---------------------------------------------------------------------------
extern "C" void kernel_launch(void* const* d_in, const int* in_sizes, int n_in,
                              void* d_out, int out_size, void* d_ws, size_t ws_size,
                              hipStream_t stream) {
  const float* x          = (const float*)d_in[0];
  const float* emb        = (const float*)d_in[1];
  const float* ema_count  = (const float*)d_in[2];
  const float* ema_weight = (const float*)d_in[3];

  const int B = in_sizes[0] / (N_CB * D_DIM * L_LAT);   // 2048
  const int T = B * L_LAT;                              // 32768

  // d_out layout: z_q | loss | perplexity | indices | enc_zq | new_embedding
  //               | new_ema_count | new_ema_weight
  float* out = (float*)d_out;
  const size_t zq_sz  = (size_t)B * N_CB * D_DIM * L_LAT;
  const size_t idx_sz = (size_t)B * N_CB * L_LAT;
  float* zq    = out;
  float* loss  = zq + zq_sz;
  float* perp  = loss + 1;
  float* idxo  = perp + 1;
  float* enczq = idxo + idx_sz;
  float* embo  = enczq + zq_sz;
  float* cnto  = embo + (size_t)N_CB * M_CODES * D_DIM;
  float* ewo   = cnto + (size_t)N_CB * M_CODES;

  float* cnt_ws = (float*)d_ws;                 // N*M raw counts
  float* sse_ws = cnt_ws + N_CB * M_CODES;      // 1 float

  {
    int total = N_CB * M_CODES * D_DIM;
    vq_init_kernel<<<(total + 255) / 256, 256, 0, stream>>>(
        ema_weight, ewo, cnt_ws, sse_ws, perp);
  }
  {
    const size_t smem = (size_t)CHUNK_M * ES_STRIDE * 2   // Es (f16)
                      + (size_t)CHUNK_M * 4               // esq
                      + (size_t)8 * 16 * XS_STRIDE * 4;   // Xs
    const int tgroups = T / 128;
    vq_main_kernel<<<N_CB * tgroups, 256, smem, stream>>>(
        x, emb, zq, enczq, idxo, cnt_ws, ewo, sse_ws, B);
  }
  vq_finalize_kernel<<<N_CB, 256, 0, stream>>>(
      ema_count, cnt_ws, ewo, cnto, embo, perp, loss, sse_ws, T);
}
